// SimpleAttentionBlock_26929444946383
// MI455X (gfx1250) — matrix operand order repaired and checked by reading, not compiled, not run on hardware
//
#include <hip/hip_runtime.h>
#include <hip/hip_bf16.h>

typedef __bf16 bf16_t;
typedef __attribute__((ext_vector_type(16))) __bf16 v16bf;
typedef __attribute__((ext_vector_type(8)))  float  v8f;

static_assert(sizeof(v16bf) == 32, "v16bf must be 8 VGPRs");
static_assert(sizeof(v8f)   == 32, "v8f must be 8 VGPRs");

#define DMODEL 1024
#define SLEN   2048
#define NBATCH 4
#define DFF    4096
#define MTOT   (NBATCH * SLEN)   // 8192 rows total

// GEMM tiling: 256 threads = 8 waves; block tile 128x64; wave tile 32x32 (4 WMMAs)
#define BM 128
#define BN 64
#define KT 32
#define APAD 40   // bf16 elements per LDS row (32 data + 8 pad, keeps 16B alignment)
#define BPAD 40

// CDNA5 async global->LDS staging (ASYNCcnt), guarded so compile never regresses.
#if defined(__gfx1250__) && __has_builtin(__builtin_amdgcn_global_load_async_to_lds_b128)
#define ASYNC_LDS 1
// Payload type per the compiler's own diagnostic: int vector_size(16) *
typedef int async_v4i __attribute__((vector_size(16)));
#else
#define ASYNC_LDS 0
#endif

__device__ __forceinline__ void wait_async0() {
#if defined(__gfx1250__)
#if __has_builtin(__builtin_amdgcn_s_wait_asynccnt)
  __builtin_amdgcn_s_wait_asynccnt(0);
#else
  asm volatile("s_wait_asynccnt 0x0" ::: "memory");
#endif
#endif
}

__device__ __forceinline__ v16bf pack16(uint4 lo, uint4 hi) {
  union { uint4 q[2]; v16bf v; } u;
  u.q[0] = lo; u.q[1] = hi;
  return u.v;
}

// Stage one 16B chunk: async DMA into LDS when available, else VGPR round trip.
__device__ __forceinline__ void stage_chunk(const bf16_t* __restrict__ gp,
                                            bf16_t* __restrict__ lp,
                                            bool prefetch_next) {
#if ASYNC_LDS
  (void)prefetch_next;
  __builtin_amdgcn_global_load_async_to_lds_b128(
      (async_v4i*)gp, (async_v4i*)lp, 0, 0);
#else
  if (prefetch_next) __builtin_prefetch((const void*)(gp + KT), 0, 3);
  *(uint4*)lp = *(const uint4*)gp;
#endif
}

// Stage one KT-wide tile of A (BM x KT) and Bt (BN x KT) into an LDS buffer.
__device__ __forceinline__ void stage_tile(const bf16_t* __restrict__ Ab,
                                           const bf16_t* __restrict__ Bb,
                                           bf16_t* __restrict__ sa,
                                           bf16_t* __restrict__ sb,
                                           int m0, int n0, int K, int k0,
                                           int tid, bool more) {
  // A tile: 128x32 bf16 = 512 16B chunks, 2 per thread
  #pragma unroll
  for (int c = 0; c < 2; ++c) {
    int idx = tid + c * 256;           // 0..511
    int r = idx >> 2;
    int cc = (idx & 3) * 8;
    stage_chunk(Ab + (long)(m0 + r) * K + k0 + cc, sa + r * APAD + cc, more);
  }
  // Bt tile: 64x32 bf16 = 256 16B chunks, 1 per thread
  {
    int r = tid >> 2;
    int cc = (tid & 3) * 8;
    stage_chunk(Bb + (long)(n0 + r) * K + k0 + cc, sb + r * BPAD + cc, more);
  }
}

// ---------------------------------------------------------------------------
// Generic bf16 WMMA GEMM:  C[M,N] = A[M,K] (row-major, ld=K)
//                               x  Bt[N,K] (row-major, i.e. B column-major)
// Epilogues:
//   EPI=0 : f32 out, * scale               (attention scores; causal tile skip)
//   EPI=1 : bf16 out                       (Q, K projections)
//   EPI=2 : bf16 out, per-batch transposed (V projection -> V^T [B][D][S])
//   EPI=3 : bias + exact GELU -> bf16      (FFN1)
//   EPI=4 : (+bias) + residual -> f32      (attn-out + x ; FFN2 + h)
// ---------------------------------------------------------------------------
template <int EPI>
__global__ __launch_bounds__(256) void gemm_wmma(
    const bf16_t* __restrict__ A, const bf16_t* __restrict__ Bt,
    void* __restrict__ Cout,
    const float* __restrict__ bias, const float* __restrict__ residual,
    int M, int N, int K,
    long sAb, long sBb, long sCb, long sRb,   // per-batch strides (elements)
    float scale, int causal)
{
  __shared__ bf16_t sA[2][BM * APAD];
  __shared__ bf16_t sB[2][BN * BPAD];

  const int bz = blockIdx.z;
  const bf16_t* Ab = A + (long)bz * sAb;
  const bf16_t* Bb = Bt + (long)bz * sBb;

  const int m0 = blockIdx.y * BM;
  const int n0 = blockIdx.x * BN;
  if (causal && n0 >= m0 + BM) return;   // tile entirely above the diagonal

  const int tid  = threadIdx.x;
  const int lane = tid & 31;
  const int wid  = tid >> 5;     // 0..7
  const int wm   = wid & 3;      // 4 wave rows  (x32)
  const int wn   = wid >> 2;     // 2 wave cols  (x32)
  const int half = lane >> 4;    // hi/lo 16 lanes
  const int l16  = lane & 15;

  v8f acc[2][2] = {};

  const int nt = K / KT;
  // Prologue: stage tile 0 into buffer 0
  stage_tile(Ab, Bb, sA[0], sB[0], m0, n0, K, 0, tid, nt > 1);

  int cur = 0;
  for (int t = 0; t < nt; ++t) {
#if ASYNC_LDS
    wait_async0();               // our tile-t async writes have landed in LDS
#endif
    __syncthreads();             // all waves' writes visible; prev compute done

    // Kick off the next tile into the other buffer while we compute this one.
    if (t + 1 < nt)
      stage_tile(Ab, Bb, sA[cur ^ 1], sB[cur ^ 1], m0, n0, K, (t + 1) * KT,
                 tid, t + 2 < nt);

    // ---- build fragments per the CDNA5 16-bit WMMA VGPR layouts ----
    const bf16_t* sa = sA[cur];
    const bf16_t* sb = sB[cur];
    v16bf afrag[2], bfrag[2];
    #pragma unroll
    for (int sm = 0; sm < 2; ++sm) {
      int row = wm * 32 + sm * 16 + l16;
      uint4 lo = *(const uint4*)(&sa[row * APAD + 8 * half]);
      uint4 hi = *(const uint4*)(&sa[row * APAD + 16 + 8 * half]);
      afrag[sm] = pack16(lo, hi);
    }
    #pragma unroll
    for (int sn = 0; sn < 2; ++sn) {
      int col = wn * 32 + sn * 16 + l16;
      uint4 lo = *(const uint4*)(&sb[col * BPAD + 8 * half]);
      uint4 hi = *(const uint4*)(&sb[col * BPAD + 16 + 8 * half]);
      bfrag[sn] = pack16(lo, hi);
    }

    #pragma unroll
    for (int sm = 0; sm < 2; ++sm)
      #pragma unroll
      for (int sn = 0; sn < 2; ++sn)
        acc[sm][sn] = __builtin_amdgcn_wmma_f32_16x16x32_bf16(
            false, afrag[sm], false, bfrag[sn], (short)0, acc[sm][sn],
            false, false);
    cur ^= 1;
  }

  // ---- epilogue: C layout VGPR r, lane l -> row r+8*(l>>4), col l&15 ----
  #pragma unroll
  for (int sn = 0; sn < 2; ++sn) {
    int col = n0 + wn * 32 + sn * 16 + l16;
    #pragma unroll
    for (int sm = 0; sm < 2; ++sm) {
      int rowBase = m0 + wm * 32 + sm * 16 + 8 * half;
      v8f c = acc[sm][sn];
      #pragma unroll
      for (int r = 0; r < 8; ++r) {
        int row = rowBase + r;
        float v = c[r];
        if (EPI == 0) {
          ((float*)Cout)[sCb * bz + (long)row * N + col] = v * scale;
        } else if (EPI == 1) {
          ((bf16_t*)Cout)[sCb * bz + (long)row * N + col] = (bf16_t)v;
        } else if (EPI == 2) {
          int b = row >> 11;           // row / SLEN
          int s = row & (SLEN - 1);
          ((bf16_t*)Cout)[((long)(b * DMODEL + col) << 11) + s] = (bf16_t)v;
        } else if (EPI == 3) {
          float y = v + bias[col];
          y = 0.5f * y * (1.0f + erff(y * 0.7071067811865476f));  // exact GELU
          ((bf16_t*)Cout)[(long)row * N + col] = (bf16_t)y;
        } else {
          float y = v;
          if (bias) y += bias[col];
          y += residual[sRb * bz + (long)row * N + col];
          ((float*)Cout)[sCb * bz + (long)row * N + col] = y;
        }
      }
    }
  }
}

// ---------------------------------------------------------------------------
// Elementwise helpers
// ---------------------------------------------------------------------------
__global__ __launch_bounds__(256) void f32_to_bf16(
    const float* __restrict__ in, bf16_t* __restrict__ out, long n)
{
  long i = ((long)blockIdx.x * 256 + threadIdx.x) * 4;
  if (i + 3 < n) {
    float4 v = *(const float4*)(in + i);
    out[i + 0] = (bf16_t)v.x;
    out[i + 1] = (bf16_t)v.y;
    out[i + 2] = (bf16_t)v.z;
    out[i + 3] = (bf16_t)v.w;
  }
}

// out[n*K + k] = (bf16) in[k*N + n]   (weight [K,N] -> W^T [N,K] in bf16)
__global__ __launch_bounds__(256) void transpose_bf16(
    const float* __restrict__ in, bf16_t* __restrict__ out, int K, int N)
{
  long i = (long)blockIdx.x * 256 + threadIdx.x;
  long k = i / N;
  long n = i - k * N;
  if (k < K) out[n * K + k] = (bf16_t)in[i];
}

__device__ __forceinline__ float block_reduce_sum(float v, float* red) {
  int tid = threadIdx.x;
  red[tid] = v; __syncthreads();
  for (int s = 128; s > 0; s >>= 1) {
    if (tid < s) red[tid] += red[tid + s];
    __syncthreads();
  }
  float r = red[0]; __syncthreads();
  return r;
}

__device__ __forceinline__ float block_reduce_max(float v, float* red) {
  int tid = threadIdx.x;
  red[tid] = v; __syncthreads();
  for (int s = 128; s > 0; s >>= 1) {
    if (tid < s) red[tid] = fmaxf(red[tid], red[tid + s]);
    __syncthreads();
  }
  float r = red[0]; __syncthreads();
  return r;
}

// Causal softmax over one score row; writes bf16 probs, zero past the diagonal
__global__ __launch_bounds__(256) void softmax_causal(
    const float* __restrict__ scores, bf16_t* __restrict__ P)
{
  __shared__ float red[256];
  int row = blockIdx.x;                 // 0..MTOT-1
  int s = row & (SLEN - 1);
  int L = s + 1;
  const float* in = scores + (long)row * SLEN;
  bf16_t* out = P + (long)row * SLEN;
  int tid = threadIdx.x;

  float m = -3.4e38f;
  for (int j = tid; j < L; j += 256) m = fmaxf(m, in[j]);
  m = block_reduce_max(m, red);

  float sum = 0.0f;
  for (int j = tid; j < L; j += 256) sum += __expf(in[j] - m);
  sum = block_reduce_sum(sum, red);
  float inv = 1.0f / sum;

  for (int j = tid; j < SLEN; j += 256)
    out[j] = (bf16_t)((j < L) ? __expf(in[j] - m) * inv : 0.0f);
}

// LayerNorm over a row of 1024; optional bf16 copy of the output
__global__ __launch_bounds__(256) void layernorm_k(
    const float* __restrict__ in, const float* __restrict__ g,
    const float* __restrict__ b, float* __restrict__ out_f32,
    bf16_t* __restrict__ out_bf16)
{
  __shared__ float red[256];
  long row = blockIdx.x;
  int tid = threadIdx.x;
  const float* xr = in + row * DMODEL;
  float4 xv = ((const float4*)xr)[tid];

  float s = xv.x + xv.y + xv.z + xv.w;
  s = block_reduce_sum(s, red);
  float mu = s * (1.0f / DMODEL);

  float dx0 = xv.x - mu, dx1 = xv.y - mu, dx2 = xv.z - mu, dx3 = xv.w - mu;
  float q = dx0 * dx0 + dx1 * dx1 + dx2 * dx2 + dx3 * dx3;
  q = block_reduce_sum(q, red);
  float rstd = rsqrtf(q * (1.0f / DMODEL) + 1e-5f);

  float4 gv = ((const float4*)g)[tid];
  float4 bv = ((const float4*)b)[tid];
  float4 y;
  y.x = dx0 * rstd * gv.x + bv.x;
  y.y = dx1 * rstd * gv.y + bv.y;
  y.z = dx2 * rstd * gv.z + bv.z;
  y.w = dx3 * rstd * gv.w + bv.w;
  ((float4*)(out_f32 + row * DMODEL))[tid] = y;
  if (out_bf16) {
    bf16_t* o = out_bf16 + row * DMODEL + tid * 4;
    o[0] = (bf16_t)y.x; o[1] = (bf16_t)y.y; o[2] = (bf16_t)y.z; o[3] = (bf16_t)y.w;
  }
}

// ---------------------------------------------------------------------------
extern "C" void kernel_launch(void* const* d_in, const int* in_sizes, int n_in,
                              void* d_out, int out_size, void* d_ws, size_t ws_size,
                              hipStream_t stream) {
  (void)in_sizes; (void)n_in; (void)out_size; (void)ws_size;
  const float* x    = (const float*)d_in[0];
  const float* Wq   = (const float*)d_in[1];
  const float* Wk   = (const float*)d_in[2];
  const float* Wv   = (const float*)d_in[3];
  const float* ln1g = (const float*)d_in[4];
  const float* ln1b = (const float*)d_in[5];
  const float* ln2g = (const float*)d_in[6];
  const float* ln2b = (const float*)d_in[7];
  const float* ff1w = (const float*)d_in[8];
  const float* ff1b = (const float*)d_in[9];
  const float* ff2w = (const float*)d_in[10];
  const float* ff2b = (const float*)d_in[11];

  char* ws = (char*)d_ws;
  size_t off = 0;
  auto take = [&](size_t bytes) -> char* {
    char* p = ws + off;
    off += (bytes + 255) & ~(size_t)255;
    return p;
  };

  bf16_t* xb  = (bf16_t*)take((size_t)MTOT * DMODEL * 2);
  bf16_t* wqT = (bf16_t*)take((size_t)DMODEL * DMODEL * 2);
  bf16_t* wkT = (bf16_t*)take((size_t)DMODEL * DMODEL * 2);
  bf16_t* wvT = (bf16_t*)take((size_t)DMODEL * DMODEL * 2);
  bf16_t* f1T = (bf16_t*)take((size_t)DMODEL * DFF * 2);
  bf16_t* f2T = (bf16_t*)take((size_t)DFF * DMODEL * 2);
  bf16_t* Qb  = (bf16_t*)take((size_t)MTOT * DMODEL * 2);
  bf16_t* Kb  = (bf16_t*)take((size_t)MTOT * DMODEL * 2);
  bf16_t* Vt  = (bf16_t*)take((size_t)MTOT * DMODEL * 2);
  // scores f32 (67MB) later reused as ff1 output bf16 (8192*4096*2 = 67MB)
  float*  scores = (float*)take((size_t)NBATCH * SLEN * SLEN * 4);
  bf16_t* ff1o   = (bf16_t*)scores;
  // P bf16 (33.5MB) later reused as h f32 (8192*1024*4 = 33.5MB)
  bf16_t* Pb = (bf16_t*)take((size_t)NBATCH * SLEN * SLEN * 2);
  float*  hf = (float*)Pb;
  float*  tmp = (float*)take((size_t)MTOT * DMODEL * 4);   // pre-LN1, later pre-LN2
  bf16_t* hb  = (bf16_t*)take((size_t)MTOT * DMODEL * 2);

  dim3 blk(256);

  // Precision conversion + weight transposes
  f32_to_bf16<<<(MTOT * DMODEL) / 1024, blk, 0, stream>>>(x, xb, (long)MTOT * DMODEL);
  transpose_bf16<<<(DMODEL * DMODEL) / 256, blk, 0, stream>>>(Wq, wqT, DMODEL, DMODEL);
  transpose_bf16<<<(DMODEL * DMODEL) / 256, blk, 0, stream>>>(Wk, wkT, DMODEL, DMODEL);
  transpose_bf16<<<(DMODEL * DMODEL) / 256, blk, 0, stream>>>(Wv, wvT, DMODEL, DMODEL);
  transpose_bf16<<<(DMODEL * DFF) / 256, blk, 0, stream>>>(ff1w, f1T, DMODEL, DFF);
  transpose_bf16<<<(DFF * DMODEL) / 256, blk, 0, stream>>>(ff2w, f2T, DFF, DMODEL);

  // Q, K (bf16), V (bf16, transposed per batch)
  dim3 gq(DMODEL / BN, MTOT / BM, 1);
  gemm_wmma<1><<<gq, blk, 0, stream>>>(xb, wqT, Qb, nullptr, nullptr,
      MTOT, DMODEL, DMODEL, 0, 0, 0, 0, 1.0f, 0);
  gemm_wmma<1><<<gq, blk, 0, stream>>>(xb, wkT, Kb, nullptr, nullptr,
      MTOT, DMODEL, DMODEL, 0, 0, 0, 0, 1.0f, 0);
  gemm_wmma<2><<<gq, blk, 0, stream>>>(xb, wvT, Vt, nullptr, nullptr,
      MTOT, DMODEL, DMODEL, 0, 0, 0, 0, 1.0f, 0);

  // scores = Q K^T / sqrt(D), causal tile skip (K rows are already [n][k])
  dim3 gs(SLEN / BN, SLEN / BM, NBATCH);
  gemm_wmma<0><<<gs, blk, 0, stream>>>(Qb, Kb, scores, nullptr, nullptr,
      SLEN, SLEN, DMODEL,
      (long)SLEN * DMODEL, (long)SLEN * DMODEL, (long)SLEN * SLEN, 0,
      0.03125f, 1);

  softmax_causal<<<MTOT, blk, 0, stream>>>(scores, Pb);

  // attn_out + x  (P is zero-padded past the diagonal, V^T is [D][S] per batch)
  dim3 ga(DMODEL / BN, SLEN / BM, NBATCH);
  gemm_wmma<4><<<ga, blk, 0, stream>>>(Pb, Vt, tmp, nullptr, x,
      SLEN, DMODEL, SLEN,
      (long)SLEN * SLEN, (long)DMODEL * SLEN, (long)SLEN * DMODEL,
      (long)SLEN * DMODEL, 1.0f, 0);

  layernorm_k<<<MTOT, blk, 0, stream>>>(tmp, ln1g, ln1b, hf, hb);

  // FFN1: bias + exact GELU -> bf16
  dim3 g1(DFF / BN, MTOT / BM, 1);
  gemm_wmma<3><<<g1, blk, 0, stream>>>(hb, f1T, ff1o, ff1b, nullptr,
      MTOT, DFF, DMODEL, 0, 0, 0, 0, 1.0f, 0);

  // FFN2: bias + residual(h) -> f32
  dim3 g2(DMODEL / BN, MTOT / BM, 1);
  gemm_wmma<4><<<g2, blk, 0, stream>>>(ff1o, f2T, tmp, ff2b, hf,
      MTOT, DMODEL, DFF, 0, 0, 0, 0, 1.0f, 0);

  layernorm_k<<<MTOT, blk, 0, stream>>>(tmp, ln2g, ln2b, (float*)d_out, nullptr);
}
